// Recurrent_53472342835370
// MI455X (gfx1250) — matrix operand order repaired and checked by reading, not compiled, not run on hardware
//
#include <hip/hip_runtime.h>
#include <math.h>

// ---- problem constants -------------------------------------------------
#define Cc     128          // hidden / input channels
#define SEQ    8
#define BB     4            // batch (32 / SEQ)
#define HH     64
#define WW     64
#define HW     (HH*WW)      // 4096
#define CHS    264          // LDS channel stride (halfs), padded (528 B, 16B-aligned)
#define NPIX   198          // 3 rows * 66 cols staged per full-row tile
#define LZS    65           // z LDS row stride (floats)

typedef __attribute__((ext_vector_type(16))) _Float16 v16h;
typedef __attribute__((ext_vector_type(8)))  _Float16 v8h;
typedef __attribute__((ext_vector_type(8)))  float    v8f;

// ------------------------------------------------------------------------
// Weight swizzle: OIHW fp32 -> f16 in exact WMMA A-fragment order.
// Linear layout (halfs): (((tap*8 + kc)*32 + mtile)*32 + lane)*16 + j
//   lane<16 : M = mtile*16 + lane,      K(j) = {0..7, 16..23}  + kc*32
//   lane>=16: M = mtile*16 + lane-16,   K(j) = {8..15, 24..31} + kc*32
// ------------------------------------------------------------------------
__global__ void convlstm_prep_weights(const float* __restrict__ w,
                                      _Float16* __restrict__ wbuf) {
    int idx = blockIdx.x * 256 + threadIdx.x;
    const int total = 9 * 8 * 32 * 32 * 16;   // 1,179,648
    if (idx >= total) return;
    int j   = idx & 15;
    int l   = (idx >> 4) & 31;
    int mt  = (idx >> 9) & 31;
    int kc  = (idx >> 14) & 7;
    int tap = idx >> 17;
    int m   = mt * 16 + (l & 15);
    int k0  = (l < 16) ? 0 : 8;
    int kl  = (j < 8) ? (k0 + j) : (16 + k0 + (j - 8));
    int i   = kc * 32 + kl;                   // 0..255 = [x ; h] channels
    int kh  = tap / 3, kw = tap % 3;
    float v = w[(((size_t)m * 256 + i) * 3 + kh) * 3 + kw];
    wbuf[idx] = (_Float16)v;
}

// ------------------------------------------------------------------------
// One LSTM timestep: implicit-GEMM 3x3 conv (WMMA f16) + gates + BN/ReLU.
// Grid: (y=64, b=4), block 256 (8 waves). Tile = full row (64 px) x 512 ch.
// ------------------------------------------------------------------------
__global__ __launch_bounds__(256)
void convlstm_step(const float* __restrict__ x,
                   const _Float16* __restrict__ wbuf,
                   const float* __restrict__ bias,
                   const float* __restrict__ gamma,
                   const float* __restrict__ beta,
                   const float* __restrict__ mean,
                   const float* __restrict__ var,
                   const float* __restrict__ h_prev,
                   float* __restrict__ h_next,
                   float* __restrict__ c_ws,
                   float* __restrict__ out,
                   int t) {
    __shared__ __align__(16) char smem[512 * LZS * 4];  // 133,120 B (>= staging 104,544 B)
    _Float16* lin = (_Float16*)smem;                    // [NPIX][CHS] f16 activations
    float*    lz  = (float*)smem;                       // [512][LZS] f32 z (reused)

    const int tid  = threadIdx.x;
    const int lane = tid & 31;
    const int wv   = tid >> 5;          // wave id 0..7 -> out-channels [64*wv, 64*wv+64)
    const int y    = blockIdx.x;
    const int b    = blockIdx.y;
    const int bt   = b * SEQ + t;

    // ---- stage input patch: 3 rows x 66 cols x 256 ch, fp32 -> f16 -----
    for (int c = wv; c < 256; c += 8) {
        for (int p = lane; p < NPIX; p += 32) {
            int r  = p / 66, cl = p - r * 66;
            int gy = y - 1 + r;
            int gx = cl - 1;
            float v = 0.0f;
            if (gy >= 0 && gy < HH && gx >= 0 && gx < WW) {
                if (c < Cc)
                    v = x[(((size_t)bt * Cc + c) * HH + gy) * WW + gx];
                else if (t > 0)
                    v = h_prev[(((size_t)b * Cc + (c - Cc)) * HH + gy) * WW + gx];
            }
            lin[p * CHS + c] = (_Float16)v;
        }
    }
    __syncthreads();

    // ---- implicit GEMM: 512 x 64 output tile, K = 2304 -----------------
    const int  l16 = lane & 15;
    const bool hi  = lane >= 16;
    v8f acc[4][4];                                      // [mt][nt]
    #pragma unroll
    for (int a = 0; a < 4; ++a)
        #pragma unroll
        for (int n = 0; n < 4; ++n)
            acc[a][n] = (v8f){0.f,0.f,0.f,0.f,0.f,0.f,0.f,0.f};

    // per-lane weight-fragment pointer; advances by a constant per k-step
    const v16h* wp = (const v16h*)wbuf + ((size_t)(wv * 4) * 32 + lane);

    for (int tap = 0; tap < 9; ++tap) {
        const int kh = tap / 3, kw = tap % 3;
        const _Float16* lbase = lin + (kh * 66 + kw) * CHS + (hi ? 8 : 0);
        for (int kc = 0; kc < 8; ++kc) {
            // A fragments for this wave's 4 m-tiles (immediate-offset loads)
            v16h afr[4];
            #pragma unroll
            for (int mt = 0; mt < 4; ++mt)
                afr[mt] = wp[mt * 32];
            // prefetch next k-steps' weights: locality 3 -> WGP-scope
            // (fills all cache levels; SYS-scope would stop at GL2)
            __builtin_prefetch((const void*)(wp + 2 * 1024), 0, 3);
            wp += 1024;                                 // next (tap,kc) chunk

            #pragma unroll
            for (int nt = 0; nt < 4; ++nt) {
                const _Float16* bp = lbase + (nt * 16 + l16) * CHS + kc * 32;
                v8h lo  = *(const v8h*)bp;
                v8h hh2 = *(const v8h*)(bp + 16);
                v16h bfr = __builtin_shufflevector(lo, hh2,
                           0,1,2,3,4,5,6,7,8,9,10,11,12,13,14,15);
                #pragma unroll
                for (int mt = 0; mt < 4; ++mt)
                    acc[mt][nt] = __builtin_amdgcn_wmma_f32_16x16x32_f16(
                        false, afr[mt], false, bfr, (short)0, acc[mt][nt],
                        false, false);
            }
        }
    }
    __syncthreads();                                    // done reading lin

    // ---- spill z (+bias) to LDS for cross-wave gate gather -------------
    #pragma unroll
    for (int mt = 0; mt < 4; ++mt)
        #pragma unroll
        for (int nt = 0; nt < 4; ++nt)
            #pragma unroll
            for (int v = 0; v < 8; ++v) {
                int ch  = wv * 64 + mt * 16 + v + (hi ? 8 : 0);
                int pix = nt * 16 + l16;
                lz[ch * LZS + pix] = acc[mt][nt][v] + bias[ch];
            }
    __syncthreads();

    // ---- LSTM gates + BN(eval) + ReLU ----------------------------------
    for (int it = 0; it < 32; ++it) {
        int idx = it * 256 + tid;                       // 8192 = 128 ch * 64 pix
        int ch  = idx >> 6;
        int pix = idx & 63;
        float zi = lz[ch * LZS + pix];
        float zf = lz[(ch + 128) * LZS + pix];
        float zo = lz[(ch + 256) * LZS + pix];
        float zg = lz[(ch + 384) * LZS + pix];

        size_t soff = (((size_t)b * Cc + ch) * HH + y) * WW + pix;
        float cp = (t > 0) ? c_ws[soff] : 0.0f;

        float ig = 1.0f / (1.0f + __expf(-zi));
        float fg = 1.0f / (1.0f + __expf(-zf));
        float og = 1.0f / (1.0f + __expf(-zo));
        float cn = fg * cp + ig * tanhf(zg);
        float hn = og * tanhf(cn);

        c_ws[soff]   = cn;
        h_next[soff] = hn;

        float inv = rsqrtf(var[ch] + 1e-5f);
        float o   = (hn - mean[ch]) * (inv * gamma[ch]) + beta[ch];
        out[(((size_t)bt * Cc + ch) * HH + y) * WW + pix] = fmaxf(o, 0.0f);
    }
}

// ------------------------------------------------------------------------
extern "C" void kernel_launch(void* const* d_in, const int* in_sizes, int n_in,
                              void* d_out, int out_size, void* d_ws, size_t ws_size,
                              hipStream_t stream) {
    (void)in_sizes; (void)n_in; (void)out_size; (void)ws_size;
    const float* x      = (const float*)d_in[0];
    const float* w_conv = (const float*)d_in[1];
    const float* b_conv = (const float*)d_in[2];
    const float* gamma  = (const float*)d_in[3];
    const float* beta   = (const float*)d_in[4];
    const float* rmean  = (const float*)d_in[5];
    const float* rvar   = (const float*)d_in[6];
    float* out = (float*)d_out;

    // workspace layout
    const size_t STATE = (size_t)BB * Cc * HW;          // 2,097,152 floats (8 MB)
    float* h0  = (float*)d_ws;
    float* h1  = h0 + STATE;
    float* cws = h1 + STATE;
    _Float16* wbuf = (_Float16*)(cws + STATE);          // 1,179,648 halfs

    // swizzle weights into WMMA A-fragment order
    convlstm_prep_weights<<<4608, 256, 0, stream>>>(w_conv, wbuf);

    // 8 sequential timesteps, h ping-pong (halo reads), c in-place
    dim3 grid(HH, BB);
    for (int t = 0; t < SEQ; ++t) {
        float* hp = (t & 1) ? h1 : h0;
        float* hn = (t & 1) ? h0 : h1;
        convlstm_step<<<grid, 256, 0, stream>>>(
            x, wbuf, b_conv, gamma, beta, rmean, rvar,
            hp, hn, cws, out, t);
    }
}
